// VQ_VAE_56925496541525
// MI455X (gfx1250) — matrix operand order repaired and checked by reading, not compiled, not run on hardware
//
#include <hip/hip_runtime.h>
#include <hip/hip_bf16.h>
#include <math.h>

// ---------------------------------------------------------------------------
// VQ-VAE forward for gfx1250 (MI455X). bf16 WMMA (f32 accum) for all the
// GEMM-shaped stages: conv2 (K=256), conv3 (K=512), VQ distance (K=64),
// convT1 (K=256, 4-parity decomposition), convT2 (K=128).
// Deterministic two-stage reductions for BN stats and the VQ loss.
// ---------------------------------------------------------------------------

typedef __attribute__((ext_vector_type(16))) __bf16 v16bf;
typedef __attribute__((ext_vector_type(8)))  float  v8f;

#define BN_EPS 1e-5f
#define SB 64   // stat sub-blocks per channel (deterministic 2-stage reduce)

// K-index inside a 16-bit A fragment (16x32, wave32):
// lanes 0-15: v0..v3 -> K0..7, v4..v7 -> K16..23 ; lanes 16-31: +8 / +8
__device__ __forceinline__ int klocA(int i, int half) {
  int p = i >> 1, sub = i & 1;
  return sub + 2 * (p & 3) + ((p >= 4) ? 16 : 0) + 8 * half;
}

// ------------------------- conv1: 3->16, stride 2, pad 1, direct ------------
__global__ __launch_bounds__(256) void conv1_direct(
    const float* __restrict__ x, const float* __restrict__ w,
    const float* __restrict__ bias, float* __restrict__ y) {
  // x:(64,3,256,256) w:(16,3,4,4) y:(64,16,128,128)
  int i  = blockIdx.x * 256 + threadIdx.x;
  int wo = i & 127, ho = (i >> 7) & 127, co = (i >> 14) & 15, n = i >> 18;
  float acc = bias[co];
  for (int ci = 0; ci < 3; ++ci)
    for (int kh = 0; kh < 4; ++kh) {
      int hi = 2 * ho + kh - 1;
      if (hi < 0 || hi >= 256) continue;
      for (int kw = 0; kw < 4; ++kw) {
        int wi = 2 * wo + kw - 1;
        if (wi < 0 || wi >= 256) continue;
        acc += x[((n * 3 + ci) * 256 + hi) * 256 + wi] *
               w[((co * 3 + ci) * 4 + kh) * 4 + kw];
      }
    }
  y[i] = acc;
}

// ------------------- BN stats: deterministic two-stage ----------------------
__global__ __launch_bounds__(256) void bn_stats_part(
    const float* __restrict__ x, float* __restrict__ psum,
    float* __restrict__ psq, int C, int HW, int NIMG) {
  int c = blockIdx.x / SB, b = blockIdx.x % SB;
  long long cnt = (long long)NIMG * HW;
  long long chunk = cnt / SB;            // all layer counts divisible by SB
  long long i0 = (long long)b * chunk;
  float s = 0.f, q = 0.f;
  for (long long i = i0 + threadIdx.x; i < i0 + chunk; i += 256) {
    int n = (int)(i / HW), sp = (int)(i % HW);
    float v = x[((long long)(n * C + c)) * HW + sp];
    s += v; q += v * v;
  }
  __shared__ float ls[256], lq[256];
  ls[threadIdx.x] = s; lq[threadIdx.x] = q;
  __syncthreads();
  for (int o = 128; o > 0; o >>= 1) {
    if ((int)threadIdx.x < o) {
      ls[threadIdx.x] += ls[threadIdx.x + o];
      lq[threadIdx.x] += lq[threadIdx.x + o];
    }
    __syncthreads();
  }
  if (threadIdx.x == 0) { psum[blockIdx.x] = ls[0]; psq[blockIdx.x] = lq[0]; }
}

__global__ __launch_bounds__(64) void bn_stats_final(
    const float* __restrict__ psum, const float* __restrict__ psq,
    float* __restrict__ stats, int C, int HW, int NIMG) {
  __shared__ float ls[SB], lq[SB];
  int c = blockIdx.x;
  ls[threadIdx.x] = psum[c * SB + threadIdx.x];
  lq[threadIdx.x] = psq[c * SB + threadIdx.x];
  __syncthreads();
  for (int o = SB / 2; o > 0; o >>= 1) {
    if ((int)threadIdx.x < o) {
      ls[threadIdx.x] += ls[threadIdx.x + o];
      lq[threadIdx.x] += lq[threadIdx.x + o];
    }
    __syncthreads();
  }
  if (threadIdx.x == 0) {
    float cnt  = (float)NIMG * (float)HW;
    float mean = ls[0] / cnt;
    float var  = lq[0] / cnt - mean * mean;
    stats[c]     = mean;
    stats[C + c] = rsqrtf(var + BN_EPS);
  }
}

template <bool TANH>
__global__ __launch_bounds__(256) void bn_apply(
    float* __restrict__ y, const float* __restrict__ stats,
    const float* __restrict__ g, const float* __restrict__ b,
    int C, int HW, long long total) {
  long long i = (long long)blockIdx.x * 256 + threadIdx.x;
  if (i >= total) return;
  int c = (int)((i / HW) % C);
  float v = (y[i] - stats[c]) * stats[C + c] * g[c] + b[c];
  y[i] = TANH ? tanhf(v) : fmaxf(v, 0.f);
}

// ------------- strided conv as implicit GEMM on bf16 WMMA -------------------
// M = N*Ho*Wo, N = COUT, K = CIN*16. One wave owns a 16x16 tile.
template <int CIN, int HIN, int WIN>
__global__ __launch_bounds__(256) void conv_s2_wmma(
    const float* __restrict__ x, const float* __restrict__ w,
    const float* __restrict__ bias, float* __restrict__ y, int COUT) {
  const int HOUT = HIN / 2, WOUT = WIN / 2;
  const int KTOT = CIN * 16;
  int wid  = (blockIdx.x * 256 + threadIdx.x) >> 5;
  int lane = threadIdx.x & 31;
  int half = lane >> 4, lr = lane & 15;
  int ntiles = COUT >> 4;
  int nt = wid % ntiles, mt = wid / ntiles;

  int m     = mt * 16 + lr;                    // A row owned by this lane
  int n_img = m / (HOUT * WOUT);
  int rem   = m % (HOUT * WOUT);
  int ho    = rem / WOUT, wo = rem % WOUT;
  int co    = nt * 16 + lr;                    // B column owned by this lane

  v8f acc = {};
  for (int kb = 0; kb < KTOT; kb += 32) {
    v16bf a, bfr;
#pragma unroll
    for (int i = 0; i < 16; ++i) {
      int k  = kb + klocA(i, half);
      int ci = k >> 4, r2 = k & 15, kh = r2 >> 2, kw = r2 & 3;
      int hi = 2 * ho + kh - 1, wi = 2 * wo + kw - 1;
      float v = 0.f;
      if (hi >= 0 && hi < HIN && wi >= 0 && wi < WIN)
        v = x[((long long)(n_img * CIN + ci) * HIN + hi) * WIN + wi];
      a[i] = (__bf16)v;
    }
#pragma unroll
    for (int i = 0; i < 16; ++i) {
      int k  = kb + klocA(i, half);            // B: lanes hold column, K along vector
      bfr[i] = (__bf16)w[co * KTOT + k];
    }
    acc = __builtin_amdgcn_wmma_f32_16x16x32_bf16(
        false, a, false, bfr, (short)0, acc, false, false);
  }
#pragma unroll
  for (int r = 0; r < 8; ++r) {
    int mm = mt * 16 + r + half * 8;
    int ni = mm / (HOUT * WOUT);
    int rr = mm % (HOUT * WOUT);
    y[(((long long)ni * COUT + co) * HOUT + rr / WOUT) * WOUT + (rr % WOUT)] =
        acc[r] + bias[co];
  }
}

// -------- transposed conv (stride 2, pad 1, 4x4) as implicit GEMM -----------
// Parity decomposition: outputs with (ho&1, wo&1) == (ph, pw) use the fixed
// 2x2 kernel-tap subset kh' = 1-ph+2*th, kw' = 1-pw+2*tw (th,tw in {0,1}),
// with hi = ho' + ph - th, wi = wo' + pw - tw.  M enumerates
// (n, class, ho', wo'); class chunks (HIN*WIN) are multiples of 16 so B is
// tile-uniform.  K = CIN*4, N = COUT.  Weight layout: (CIN, COUT, 4, 4).
template <int CIN, int HIN, int WIN>
__global__ __launch_bounds__(256) void convT_wmma(
    const float* __restrict__ x, const float* __restrict__ w,
    const float* __restrict__ bias, float* __restrict__ y, int COUT) {
  const int KTOT = CIN * 4;
  const int HW   = HIN * WIN;
  int wid  = (blockIdx.x * 256 + threadIdx.x) >> 5;
  int lane = threadIdx.x & 31;
  int half = lane >> 4, lr = lane & 15;
  int ntiles = COUT >> 4;
  int nt = wid % ntiles, mt = wid / ntiles;

  int cls = (mt * 16 / HW) & 3;                // tile-constant parity class
  int ph = cls >> 1, pw = cls & 1;

  int m   = mt * 16 + lr;
  int n_img = m / (4 * HW);
  int r0    = m % HW;
  int hop   = r0 / WIN, wop = r0 % WIN;        // (ho', wo') in class subgrid
  int co    = nt * 16 + lr;

  v8f acc = {};
  for (int kb = 0; kb < KTOT; kb += 32) {
    v16bf a, bfr;
#pragma unroll
    for (int i = 0; i < 16; ++i) {
      int k  = kb + klocA(i, half);
      int ci = k >> 2, r2 = k & 3, th = r2 >> 1, tw = r2 & 1;
      int hi = hop + ph - th, wi = wop + pw - tw;
      float v = 0.f;
      if (hi >= 0 && hi < HIN && wi >= 0 && wi < WIN)
        v = x[((long long)(n_img * CIN + ci) * HIN + hi) * WIN + wi];
      a[i] = (__bf16)v;
    }
#pragma unroll
    for (int i = 0; i < 16; ++i) {
      int k  = kb + klocA(i, half);
      int ci = k >> 2, r2 = k & 3, th = r2 >> 1, tw = r2 & 1;
      int khp = 1 - ph + 2 * th, kwp = 1 - pw + 2 * tw;
      bfr[i] = (__bf16)w[((ci * COUT + co) * 4 + (3 - khp)) * 4 + (3 - kwp)];
    }
    acc = __builtin_amdgcn_wmma_f32_16x16x32_bf16(
        false, a, false, bfr, (short)0, acc, false, false);
  }
#pragma unroll
  for (int r = 0; r < 8; ++r) {
    int mm = mt * 16 + r + half * 8;
    int ni = mm / (4 * HW);
    int rr = mm % HW;
    int ho = 2 * (rr / WIN) + ph, wo = 2 * (rr % WIN) + pw;
    y[(((long long)ni * COUT + co) * (2 * HIN) + ho) * (2 * WIN) + wo] =
        acc[r] + bias[co];
  }
}

// ---------------- codebook norms ‖c_k‖^2 ------------------------------------
__global__ void cnorm_kernel(const float* __restrict__ cb, float* __restrict__ cn) {
  int k = blockIdx.x * 256 + threadIdx.x;
  if (k >= 512) return;
  float s = 0.f;
  for (int d = 0; d < 64; ++d) { float v = cb[k * 64 + d]; s += v * v; }
  cn[k] = s;
}

// ---------------- VQ nearest-code search via WMMA ---------------------------
__global__ __launch_bounds__(256) void vq_argmin_wmma(
    const float* __restrict__ ze, const float* __restrict__ cb,
    const float* __restrict__ cnorm, int* __restrict__ idx) {
  int wid  = (blockIdx.x * 256 + threadIdx.x) >> 5;
  int lane = threadIdx.x & 31;
  int half = lane >> 4, lr = lane & 15;
  int m = wid * 16 + lr;
  long long base = (long long)(m >> 10) * 65536 + (m & 1023);  // NCHW gather

  v16bf a0, a1;                    // K = 0..31 and 32..63, hoisted out of N loop
#pragma unroll
  for (int i = 0; i < 16; ++i) {
    int k0 = klocA(i, half);
    a0[i] = (__bf16)ze[base + (long long)k0 * 1024];
    a1[i] = (__bf16)ze[base + (long long)(k0 + 32) * 1024];
  }

  float best[8]; int bidx[8];
#pragma unroll
  for (int r = 0; r < 8; ++r) { best[r] = 3.4e38f; bidx[r] = 0; }

  for (int nt = 0; nt < 32; ++nt) {
    int code = nt * 16 + lr;
    v16bf b0, b1;
#pragma unroll
    for (int i = 0; i < 16; ++i) {
      int k = klocA(i, half);
      b0[i] = (__bf16)cb[code * 64 + k];
      b1[i] = (__bf16)cb[code * 64 + 32 + k];
    }
    v8f acc = {};
    acc = __builtin_amdgcn_wmma_f32_16x16x32_bf16(
        false, a0, false, b0, (short)0, acc, false, false);
    acc = __builtin_amdgcn_wmma_f32_16x16x32_bf16(
        false, a1, false, b1, (short)0, acc, false, false);
    float cn = cnorm[code];
#pragma unroll
    for (int r = 0; r < 8; ++r) {
      float dist = cn - 2.f * acc[r];
      if (dist < best[r] || (dist == best[r] && code < bidx[r])) {
        best[r] = dist; bidx[r] = code;
      }
    }
  }
#pragma unroll
  for (int r = 0; r < 8; ++r) {
#pragma unroll
    for (int o = 1; o < 16; o <<= 1) {
      float ov = __shfl_xor(best[r], o, 32);
      int   oi = __shfl_xor(bidx[r], o, 32);
      if (ov < best[r] || (ov == best[r] && oi < bidx[r])) {
        best[r] = ov; bidx[r] = oi;
      }
    }
  }
  if (lr == 0) {
#pragma unroll
    for (int r = 0; r < 8; ++r) idx[wid * 16 + r + half * 8] = bidx[r];
  }
}

// ------------- gather z_q (NCHW), per-block loss partials, idx out ----------
__global__ __launch_bounds__(256) void vq_gather_loss(
    const int* __restrict__ idx, const float* __restrict__ cb,
    const float* __restrict__ ze, float* __restrict__ zq,
    float* __restrict__ partials, float* __restrict__ out_idx) {
  int e = blockIdx.x * 256 + threadIdx.x;   // 4,194,304 total
  int c = e >> 16;                          // channel-major for coalescing
  int m = e & 65535;
  long long a = (long long)(m >> 10) * 65536 + (long long)c * 1024 + (m & 1023);
  int   id = idx[m];
  float q  = cb[id * 64 + c];
  zq[a] = q;
  float d = q - ze[a];
  if (c == 0) out_idx[m] = (float)id;
  __shared__ float ls[256];
  ls[threadIdx.x] = d * d;
  __syncthreads();
  for (int o = 128; o > 0; o >>= 1) {
    if ((int)threadIdx.x < o) ls[threadIdx.x] += ls[threadIdx.x + o];
    __syncthreads();
  }
  if (threadIdx.x == 0) partials[blockIdx.x] = ls[0];
}

__global__ __launch_bounds__(256) void loss_finalize(
    const float* __restrict__ partials, float* __restrict__ out_loss) {
  __shared__ float ls[256];
  float s = 0.f;
  for (int i = threadIdx.x; i < 16384; i += 256) s += partials[i];
  ls[threadIdx.x] = s;
  __syncthreads();
  for (int o = 128; o > 0; o >>= 1) {
    if ((int)threadIdx.x < o) ls[threadIdx.x] += ls[threadIdx.x + o];
    __syncthreads();
  }
  if (threadIdx.x == 0) out_loss[0] = 2.f * ls[0] / 4194304.f;  // codebook+commit
}

// ---------------- transposed conv direct (for COUT=3 tail layer) ------------
template <int CIN, int HIN, int WIN>
__global__ __launch_bounds__(256) void convT_direct(
    const float* __restrict__ x, const float* __restrict__ w,
    const float* __restrict__ bias, float* __restrict__ y, int COUT) {
  const int HOUT = HIN * 2, WOUT = WIN * 2;
  long long i = (long long)blockIdx.x * 256 + threadIdx.x;
  int wo = (int)(i % WOUT);
  int ho = (int)((i / WOUT) % HOUT);
  int co = (int)((i / ((long long)WOUT * HOUT)) % COUT);
  int n  = (int)(i / ((long long)WOUT * HOUT * COUT));
  float acc = bias[co];
#pragma unroll
  for (int kh = 0; kh < 4; ++kh) {
    int th = ho + 1 - kh;
    if (th < 0 || (th & 1)) continue;
    int hi = th >> 1;
    if (hi >= HIN) continue;
#pragma unroll
    for (int kw = 0; kw < 4; ++kw) {
      int tw = wo + 1 - kw;
      if (tw < 0 || (tw & 1)) continue;
      int wi = tw >> 1;
      if (wi >= WIN) continue;
      const float* xp = x + ((long long)(n * CIN) * HIN + hi) * WIN + wi;
      const float* wp = w + (co * 4 + (3 - kh)) * 4 + (3 - kw);
      for (int ci = 0; ci < CIN; ++ci)
        acc += xp[(long long)ci * HIN * WIN] * wp[(long long)ci * COUT * 16];
    }
  }
  y[i] = acc;
}

// ---------------------------------------------------------------------------
extern "C" void kernel_launch(void* const* d_in, const int* in_sizes, int n_in,
                              void* d_out, int out_size, void* d_ws, size_t ws_size,
                              hipStream_t stream) {
  (void)in_sizes; (void)n_in; (void)out_size; (void)ws_size;
  const float* x    = (const float*)d_in[0];
  const float* cb   = (const float*)d_in[1];
  const float* ew1  = (const float*)d_in[2];  const float* eb1 = (const float*)d_in[3];
  const float* ew2  = (const float*)d_in[4];  const float* eb2 = (const float*)d_in[5];
  const float* ew3  = (const float*)d_in[6];  const float* eb3 = (const float*)d_in[7];
  const float* dw1  = (const float*)d_in[8];  const float* db1 = (const float*)d_in[9];
  const float* dw2  = (const float*)d_in[10]; const float* db2 = (const float*)d_in[11];
  const float* dw3  = (const float*)d_in[12]; const float* db3 = (const float*)d_in[13];
  const float* ge1  = (const float*)d_in[14]; const float* bte1 = (const float*)d_in[15];
  const float* ge2  = (const float*)d_in[16]; const float* bte2 = (const float*)d_in[17];
  const float* ge3  = (const float*)d_in[18]; const float* bte3 = (const float*)d_in[19];
  const float* gd1  = (const float*)d_in[20]; const float* btd1 = (const float*)d_in[21];
  const float* gd2  = (const float*)d_in[22]; const float* btd2 = (const float*)d_in[23];
  const float* gd3  = (const float*)d_in[24]; const float* btd3 = (const float*)d_in[25];

  float* ws    = (float*)d_ws;
  float* bufA  = ws;                       // 16,777,216 f  (h1, later g2)
  float* bufB  = bufA + 16777216;          //  8,388,608 f  (h2, later g1)
  float* bufC  = bufB + 8388608;           //  4,194,304 f  (z_e)
  float* bufD  = bufC + 4194304;           //  4,194,304 f  (z_q)
  int*   idxb  = (int*)(bufD + 4194304);   //     65,536 i
  float* cn    = (float*)(idxb + 65536);   //        512 f
  float* stats = cn + 512;                 //        128 f  (mean | invstd)
  float* psum  = stats + 128;              //      4,096 f
  float* psq   = psum + 4096;              //      4,096 f
  float* lpart = psq + 4096;               //     16,384 f

  float* z_d      = (float*)d_out;         // 12,582,912 f
  float* out_idx  = z_d + 12582912;        //     65,536 f
  float* out_loss = out_idx + 65536;       //          1 f

  // ---- encoder ----
  conv1_direct<<<65536, 256, 0, stream>>>(x, ew1, eb1, bufA);
  bn_stats_part<<<16 * SB, 256, 0, stream>>>(bufA, psum, psq, 16, 16384, 64);
  bn_stats_final<<<16, SB, 0, stream>>>(psum, psq, stats, 16, 16384, 64);
  bn_apply<false><<<65536, 256, 0, stream>>>(bufA, stats, ge1, bte1, 16, 16384, 16777216LL);

  conv_s2_wmma<16, 128, 128><<<4096, 256, 0, stream>>>(bufA, ew2, eb2, bufB, 32);
  bn_stats_part<<<32 * SB, 256, 0, stream>>>(bufB, psum, psq, 32, 4096, 64);
  bn_stats_final<<<32, SB, 0, stream>>>(psum, psq, stats, 32, 4096, 64);
  bn_apply<false><<<32768, 256, 0, stream>>>(bufB, stats, ge2, bte2, 32, 4096, 8388608LL);

  conv_s2_wmma<32, 64, 64><<<2048, 256, 0, stream>>>(bufB, ew3, eb3, bufC, 64);
  bn_stats_part<<<64 * SB, 256, 0, stream>>>(bufC, psum, psq, 64, 1024, 64);
  bn_stats_final<<<64, SB, 0, stream>>>(psum, psq, stats, 64, 1024, 64);
  bn_apply<false><<<16384, 256, 0, stream>>>(bufC, stats, ge3, bte3, 64, 1024, 4194304LL);

  // ---- vector quantization ----
  cnorm_kernel<<<2, 256, 0, stream>>>(cb, cn);
  vq_argmin_wmma<<<512, 256, 0, stream>>>(bufC, cb, cn, idxb);
  vq_gather_loss<<<16384, 256, 0, stream>>>(idxb, cb, bufC, bufD, lpart, out_idx);
  loss_finalize<<<1, 256, 0, stream>>>(lpart, out_loss);

  // ---- decoder ----
  // dT1: M = 64*4*32*32 = 262144 -> 16384 M-tiles x 2 N-tiles = 4096 blocks
  convT_wmma<64, 32, 32><<<4096, 256, 0, stream>>>(bufD, dw1, db1, bufB, 32);
  bn_stats_part<<<32 * SB, 256, 0, stream>>>(bufB, psum, psq, 32, 4096, 64);
  bn_stats_final<<<32, SB, 0, stream>>>(psum, psq, stats, 32, 4096, 64);
  bn_apply<false><<<32768, 256, 0, stream>>>(bufB, stats, gd1, btd1, 32, 4096, 8388608LL);

  // dT2: M = 64*4*64*64 = 1048576 -> 65536 M-tiles x 1 N-tile = 8192 blocks
  convT_wmma<32, 64, 64><<<8192, 256, 0, stream>>>(bufB, dw2, db2, bufA, 16);
  bn_stats_part<<<16 * SB, 256, 0, stream>>>(bufA, psum, psq, 16, 16384, 64);
  bn_stats_final<<<16, SB, 0, stream>>>(psum, psq, stats, 16, 16384, 64);
  bn_apply<false><<<65536, 256, 0, stream>>>(bufA, stats, gd2, btd2, 16, 16384, 16777216LL);

  convT_direct<16, 128, 128><<<49152, 256, 0, stream>>>(bufA, dw3, db3, z_d, 3);
  bn_stats_part<<<3 * SB, 256, 0, stream>>>(z_d, psum, psq, 3, 65536, 64);
  bn_stats_final<<<3, SB, 0, stream>>>(psum, psq, stats, 3, 65536, 64);
  bn_apply<true><<<49152, 256, 0, stream>>>(z_d, stats, gd3, btd3, 3, 65536, 12582912LL);
}
